// MultiHeadSelfAttention_47768626266402
// MI455X (gfx1250) — matrix operand order repaired and checked by reading, not compiled, not run on hardware
//
#include <hip/hip_runtime.h>
#include <hip/hip_bf16.h>

// ---------------- CDNA5 WMMA types ----------------
typedef __attribute__((ext_vector_type(16))) _Float16 v16h;
typedef __attribute__((ext_vector_type(8)))  float    v8f;

union Frag16 { v16h v; uint4 u[2]; };

// A-matrix fragment (16x32 f16, M x K), A row-major with leading dim lda.
// Layout (ISA 7.12.2): lanes 0-15 -> M=lane, slots0-7 = K k0..k0+7, slots8-15 = K k0+16..k0+23
//                      lanes16-31 -> M=lane-16, same but K base +8.
__device__ inline v16h load_afrag(const _Float16* A, int lda, int row_base, int k0, int lane) {
    int m  = row_base + (lane & 15);
    int kk = k0 + ((lane & 16) ? 8 : 0);
    const _Float16* p = A + (size_t)m * lda + kk;
    Frag16 f;
    f.u[0] = *(const uint4*)(p);
    f.u[1] = *(const uint4*)(p + 16);
    return f.v;
}

// B-matrix fragment (32x16 f16, K x N) sourced from Bt = B^T stored [N][K] row-major (ld ldb).
// Layout: lanes 0-15 -> N=lane, slots0-7 = K k0..k0+7, slots8-15 = K k0+16..k0+23 ; lanes16-31 -> N=lane-16, same but K base +8.
__device__ inline v16h load_bfrag(const _Float16* Bt, int ldb, int col_base, int k0, int lane) {
    int n  = col_base + (lane & 15);
    int kk = k0 + ((lane & 16) ? 8 : 0);
    const _Float16* p = Bt + (size_t)n * ldb + kk;
    Frag16 f;
    f.u[0] = *(const uint4*)(p);
    f.u[1] = *(const uint4*)(p + 16);
    return f.v;
}

__device__ inline v8f wmma_f16(v16h a, v16h b, v8f c) {
    // D = A(16x32) * B(32x16) + C, f32 accumulate
    return __builtin_amdgcn_wmma_f32_16x16x32_f16(false, a, false, b, (short)0, c, false, false);
}

// ---------------- conversion kernels ----------------
__global__ void cvt_f32_to_f16(const float* __restrict__ in, _Float16* __restrict__ out, int n) {
    int i = blockIdx.x * blockDim.x + threadIdx.x;
    if (i < n) out[i] = (_Float16)in[i];
}

// Wt[j][i] = (f16) W[i][j], W is dim x dim row-major
__global__ void transpose_cvt_f16(const float* __restrict__ W, _Float16* __restrict__ Wt, int dim) {
    int i = blockIdx.x * blockDim.x + threadIdx.x;
    if (i < dim * dim) {
        int r = i / dim, c = i % dim;
        Wt[(size_t)c * dim + r] = (_Float16)W[(size_t)r * dim + c];
    }
}

// ---------------- WMMA GEMM: C[M x N] = A[M x K] * Bt^T, one wave = 32x64 tile ----------------
// Per K-step: 2 A fragments + 4 B fragments -> 8 WMMAs (B reused across both row tiles).
// mode 0: C fp32 row-major [M][N]
// mode 1: C f16  [b][h][s][dk]   (head-split, for Q and K)
// mode 2: C f16  [b][h][dk][s]   (head-split + transposed, for V)
__global__ void __launch_bounds__(32)
gemm_wmma_32x64(const _Float16* __restrict__ A, const _Float16* __restrict__ Bt,
                void* __restrict__ C, int M, int N, int K, int mode) {
    int lane     = threadIdx.x;
    int row_base = blockIdx.y * 32;
    int col_base = blockIdx.x * 64;

    v8f acc[2][4] = {};
#pragma unroll 2
    for (int k = 0; k < K; k += 32) {
        v16h a0 = load_afrag(A, K, row_base,      k, lane);
        v16h a1 = load_afrag(A, K, row_base + 16, k, lane);
#pragma unroll
        for (int n = 0; n < 4; ++n) {
            v16h b = load_bfrag(Bt, K, col_base + n * 16, k, lane);
            acc[0][n] = wmma_f16(a0, b, acc[0][n]);
            acc[1][n] = wmma_f16(a1, b, acc[1][n]);
        }
    }

    // C layout: VGPR r, lanes0-15 -> (M=r, N=lane); lanes16-31 -> (M=r+8, N=lane-16)
    int nl  = lane & 15;
    int hi8 = (lane & 16) ? 8 : 0;
#pragma unroll
    for (int mt = 0; mt < 2; ++mt) {
#pragma unroll
        for (int n = 0; n < 4; ++n) {
#pragma unroll
            for (int r = 0; r < 8; ++r) {
                int   row = row_base + mt * 16 + r + hi8;
                int   col = col_base + n * 16 + nl;
                float v   = acc[mt][n][r];
                if (mode == 0) {
                    ((float*)C)[(size_t)row * N + col] = v;
                } else {
                    int b_idx = row >> 11;          // row / 2048
                    int s_idx = row & 2047;
                    int head  = col >> 6;           // col / 64
                    int j     = col & 63;
                    _Float16* Ch = (_Float16*)C;
                    if (mode == 1)
                        Ch[(((size_t)(b_idx * 16 + head) * 2048) + s_idx) * 64 + j] = (_Float16)v;
                    else
                        Ch[(((size_t)(b_idx * 16 + head) * 64) + j) * 2048 + s_idx] = (_Float16)v;
                }
            }
        }
    }
}

// ---------------- flash attention: one wave per (b,h,q-tile of 16) ----------------
// Q,K: [b*h][2048][64] f16 ; Vt: [b*h][64][2048] f16 ; attn out: [b][2048][1024] f16
__global__ void __launch_bounds__(32)
flash_attn_wmma(const _Float16* __restrict__ Q, const _Float16* __restrict__ K,
                const _Float16* __restrict__ Vt, _Float16* __restrict__ attn) {
    __shared__ __attribute__((aligned(16))) _Float16 Pld[16 * 32];

    int lane = threadIdx.x;
    int qt   = blockIdx.x & 127;   // s/16 = 128 q-tiles
    int bh   = blockIdx.x >> 7;    // 0..31
    int q_base = qt * 16;

    const _Float16* Qh = Q  + (size_t)bh * 2048 * 64;
    const _Float16* Kh = K  + (size_t)bh * 2048 * 64;
    const _Float16* Vh = Vt + (size_t)bh * 64 * 2048;

    // Q tile A-fragments for dk = 0..31 and 32..63
    v16h aq0 = load_afrag(Qh, 64, q_base, 0, lane);
    v16h aq1 = load_afrag(Qh, 64, q_base, 32, lane);

    v8f   acc[4] = {};
    float m[8], l[8];
#pragma unroll
    for (int r = 0; r < 8; ++r) { m[r] = -1e30f; l[r] = 0.0f; }

    int nl  = lane & 15;
    int hi8 = (lane & 16) ? 8 : 0;
    const float scale = 0.125f;   // 1/sqrt(64)

    for (int kb = 0; kb < q_base + 16; kb += 32) {   // uniform bound: skip fully-masked blocks
        // S tiles: keys [kb..kb+15] and [kb+16..kb+31]
        v8f c0 = {}, c1 = {};
        c0 = wmma_f16(aq0, load_bfrag(Kh, 64, kb,       0,  lane), c0);
        c0 = wmma_f16(aq1, load_bfrag(Kh, 64, kb,       32, lane), c0);
        c1 = wmma_f16(aq0, load_bfrag(Kh, 64, kb + 16,  0,  lane), c1);
        c1 = wmma_f16(aq1, load_bfrag(Kh, 64, kb + 16,  32, lane), c1);

        int key0 = kb + nl;
        int key1 = key0 + 16;
#pragma unroll
        for (int r = 0; r < 8; ++r) {
            int   row = q_base + r + hi8;
            float s0  = (key0 <= row) ? c0[r] * scale : -1e30f;
            float s1  = (key1 <= row) ? c1[r] * scale : -1e30f;

            // row-wise max across the 16 lanes holding this row (xor stays within half)
            float t = fmaxf(s0, s1);
            t = fmaxf(t, __shfl_xor(t, 1, 32));
            t = fmaxf(t, __shfl_xor(t, 2, 32));
            t = fmaxf(t, __shfl_xor(t, 4, 32));
            t = fmaxf(t, __shfl_xor(t, 8, 32));
            float mnew = fmaxf(m[r], t);
            float fac  = __expf(m[r] - mnew);
            float p0   = __expf(s0 - mnew);
            float p1   = __expf(s1 - mnew);

            float rs = p0 + p1;
            rs += __shfl_xor(rs, 1, 32);
            rs += __shfl_xor(rs, 2, 32);
            rs += __shfl_xor(rs, 4, 32);
            rs += __shfl_xor(rs, 8, 32);

            l[r] = l[r] * fac + rs;
            m[r] = mnew;
#pragma unroll
            for (int n = 0; n < 4; ++n) acc[n][r] *= fac;

            // stage P (16 x 32, row-major) in LDS for A-fragment reload
            Pld[(r + hi8) * 32 + nl]      = (_Float16)p0;
            Pld[(r + hi8) * 32 + nl + 16] = (_Float16)p1;
        }
        __syncthreads();

        v16h ap = load_afrag(Pld, 32, 0, 0, lane);   // P as 16x32 A-matrix
#pragma unroll
        for (int n = 0; n < 4; ++n)
            acc[n] = wmma_f16(ap, load_bfrag(Vh, 2048, n * 16, kb, lane), acc[n]);
        __syncthreads();
    }

    // normalize and write attn output merged back to [b][s][h*dk]
    int b_idx = bh >> 4;
    int head  = bh & 15;
#pragma unroll
    for (int r = 0; r < 8; ++r) {
        float inv = 1.0f / l[r];
        int   row = q_base + r + hi8;
        size_t base = ((size_t)b_idx * 2048 + row) * 1024 + head * 64;
#pragma unroll
        for (int n = 0; n < 4; ++n)
            attn[base + n * 16 + nl] = (_Float16)(acc[n][r] * inv);
    }
}

// ---------------- launcher ----------------
static constexpr int kQTilesPerBatch = 16 * 128;   // h * (s/16) = 2048

extern "C" void kernel_launch(void* const* d_in, const int* in_sizes, int n_in,
                              void* d_out, int out_size, void* d_ws, size_t ws_size,
                              hipStream_t stream) {
    const int B = 2, S = 2048, D = 1024;
    const int M = B * S;                     // 4096
    const float* x  = (const float*)d_in[0];
    const float* Wq = (const float*)d_in[1];
    const float* Wk = (const float*)d_in[2];
    const float* Wv = (const float*)d_in[3];
    const float* Wo = (const float*)d_in[4];
    float* out = (float*)d_out;

    char* ws = (char*)d_ws;
    const size_t MB = 1u << 20;
    _Float16* Xh   = (_Float16*)(ws + 0 * MB);    // 4096x1024  (8 MB)
    _Float16* Wqt  = (_Float16*)(ws + 8 * MB);    // 1024x1024  (2 MB) transposed
    _Float16* Wkt  = (_Float16*)(ws + 10 * MB);
    _Float16* Wvt  = (_Float16*)(ws + 12 * MB);
    _Float16* Wot  = (_Float16*)(ws + 14 * MB);
    _Float16* Qb   = (_Float16*)(ws + 16 * MB);   // [b,h,s,dk] (8 MB)
    _Float16* Kb   = (_Float16*)(ws + 24 * MB);   // [b,h,s,dk]
    _Float16* Vtb  = (_Float16*)(ws + 32 * MB);   // [b,h,dk,s]
    _Float16* Attn = (_Float16*)(ws + 40 * MB);   // [b,s,d]    (8 MB)

    // convert inputs to f16
    {
        int n = M * D;
        cvt_f32_to_f16<<<(n + 255) / 256, 256, 0, stream>>>(x, Xh, n);
        int nw = D * D;
        transpose_cvt_f16<<<(nw + 255) / 256, 256, 0, stream>>>(Wq, Wqt, D);
        transpose_cvt_f16<<<(nw + 255) / 256, 256, 0, stream>>>(Wk, Wkt, D);
        transpose_cvt_f16<<<(nw + 255) / 256, 256, 0, stream>>>(Wv, Wvt, D);
        transpose_cvt_f16<<<(nw + 255) / 256, 256, 0, stream>>>(Wo, Wot, D);
    }

    dim3 ggrid(D / 64, M / 32);   // (16, 128) -> 2048 waves per GEMM
    // projections: Q,K head-split; V head-split + transposed
    gemm_wmma_32x64<<<ggrid, 32, 0, stream>>>(Xh, Wqt, Qb,  M, D, D, 1);
    gemm_wmma_32x64<<<ggrid, 32, 0, stream>>>(Xh, Wkt, Kb,  M, D, D, 1);
    gemm_wmma_32x64<<<ggrid, 32, 0, stream>>>(Xh, Wvt, Vtb, M, D, D, 2);

    // attention: one wave per (b,h,q-tile) -> 32 * 128 = 4096 waves
    flash_attn_wmma<<<B * kQTilesPerBatch, 32, 0, stream>>>(Qb, Kb, Vtb, Attn);

    // output projection -> fp32
    gemm_wmma_32x64<<<ggrid, 32, 0, stream>>>(Attn, Wot, out, M, D, D, 0);
}
